// Generator_53223234732199
// MI455X (gfx1250) — matrix operand order repaired and checked by reading, compiled, not run on hardware
//
#include <hip/hip_runtime.h>
#include <hip/hip_bf16.h>
#include <stdint.h>

// ---------------------------------------------------------------------------
// Masked-GRU recurrence on MI455X (gfx1250), bf16 WMMA, persistent kernel.
//
// B=128, T=1024, F=256, H=512.
// - Weights repacked once per launch into v_wmma_f32_16x16x32_bf16 B-fragment
//   layout (ISA 7.12.2), then each WG stages its 88KB column slice into LDS
//   (320KB/WGP) and serves all 1023 steps from ds_load_b128.
// - Activations (xm, h) kept in A-fragment layout in global (cross-WG),
//   h fragments double-buffered across steps.
// - Recurrence state h lives entirely in registers (fixed tile ownership).
// - 2 device-wide barriers per timestep (atomic monotonic counter).
// ---------------------------------------------------------------------------

#define B_ 128
#define T_ 1024
#define F_ 256
#define H_ 512
#define NWG 32              // persistent workgroups (must all be resident)

typedef __attribute__((ext_vector_type(16))) __bf16 v16bf;
typedef __attribute__((ext_vector_type(8)))  float  v8f;

// fragment sizes (in u16 elements)
#define TILE_ELEMS 512                       // 16x32 or 32x16 bf16 tile
#define WX_TILES   (8 * 96)                  // kt(F/32) x nt(3H/16)
#define WH_TILES   (16 * 96)                 // kt(H/32) x nt(3H/16)
#define WO_TILES   (16 * 16)                 // kt(H/32) x nt(F/16)
#define HPK_TILES  (8 * 16)                  // bi(B/16) x kt(H/32)
#define XMPK_TILES (8 * 8)                   // bi(B/16) x kt(F/32)

// workspace byte offsets (256B aligned)
#define OFF_WX    0u
#define OFF_WH    786432u
#define OFF_WO    2359296u
#define OFF_HPK   2621440u                   // 2 buffers x 65536 u16
#define OFF_XMPK  2883584u
#define OFF_CNT   2949120u

// LDS layout (u16 units); dynamic shared memory, 90112 bytes total
#define SM_WX   0                            // 24 tiles (kt 0..7 x gate 0..2)
#define SM_WH   12288                        // 48 tiles (kt 0..15 x gate 0..2)
#define SM_WO   36864                        // 16 tiles (kt 0..15)
#define SM_U16  45056
#define SM_BYTES (SM_U16 * 2)

__device__ __forceinline__ unsigned short f2bf(float f) {
  unsigned u = __float_as_uint(f);
  u += 0x7FFFu + ((u >> 16) & 1u);           // round-to-nearest-even
  return (unsigned short)(u >> 16);
}

__device__ __forceinline__ float sigm(float x) {
  return 1.0f / (1.0f + __expf(-x));
}

__device__ __forceinline__ v8f wmma_bf16(v16bf a, v16bf b, v8f c) {
  return __builtin_amdgcn_wmma_f32_16x16x32_bf16(
      false, a, false, b, (short)0, c, false, false);
}

__device__ __forceinline__ v16bf load_frag_g(const unsigned short* base, int tile, int lane) {
  return *reinterpret_cast<const v16bf*>(base + tile * TILE_ELEMS + lane * 16);
}

// ---------------------------------------------------------------------------
// Weight repack: fp32 row-major -> bf16 WMMA B-fragment layout.
// B-frag mapping (16-bit, 32x16): col = lane&15, K = (e<8) ? e + 8*(lane>>4) : e + 8 + 8*(lane>>4).
// ---------------------------------------------------------------------------
__global__ void pack_weights(const float* __restrict__ Wxz, const float* __restrict__ Wxr,
                             const float* __restrict__ Wxg, const float* __restrict__ Whz,
                             const float* __restrict__ Whr, const float* __restrict__ Whg,
                             const float* __restrict__ Wout,
                             unsigned short* __restrict__ Wx,
                             unsigned short* __restrict__ Wh,
                             unsigned short* __restrict__ Wo) {
  const int NWx = WX_TILES * TILE_ELEMS;
  const int NWh = WH_TILES * TILE_ELEMS;
  const int NWo = WO_TILES * TILE_ELEMS;
  int d = blockIdx.x * blockDim.x + threadIdx.x;
  if (d < NWx) {
    int e = d & 15, lane = (d >> 4) & 31, tix = d >> 9;
    int kt = tix / 96, nt = tix % 96;
    int k = (e < 8) ? (e + ((lane >> 4) << 3)) : (e + 8 + ((lane >> 4) << 3));          // 0..31
    int n = lane & 15;
    int K = kt * 32 + k;                     // 0..255 (F)
    int gate = nt >> 5;
    int col = ((nt & 31) << 4) + n;          // 0..511 (H)
    const float* S = (gate == 0) ? Wxz : (gate == 1) ? Wxr : Wxg;
    Wx[d] = f2bf(S[K * H_ + col]);
    return;
  }
  d -= NWx;
  if (d < NWh) {
    int e = d & 15, lane = (d >> 4) & 31, tix = d >> 9;
    int kt = tix / 96, nt = tix % 96;
    int k = (e < 8) ? (e + ((lane >> 4) << 3)) : (e + 8 + ((lane >> 4) << 3));
    int n = lane & 15;
    int K = kt * 32 + k;                     // 0..511 (H)
    int gate = nt >> 5;
    int col = ((nt & 31) << 4) + n;
    const float* S = (gate == 0) ? Whz : (gate == 1) ? Whr : Whg;
    Wh[d] = f2bf(S[K * H_ + col]);
    return;
  }
  d -= NWh;
  if (d < NWo) {
    int e = d & 15, lane = (d >> 4) & 31, tix = d >> 9;
    int kt = tix >> 4, ft = tix & 15;
    int k = (e < 8) ? (e + ((lane >> 4) << 3)) : (e + 8 + ((lane >> 4) << 3));
    int n = lane & 15;
    int K = kt * 32 + k;                     // 0..511 (H)
    int f = (ft << 4) + n;                   // 0..255 (F)
    Wo[d] = f2bf(Wout[f * H_ + K]);          // W_out^T as B matrix
  }
}

// ---------------------------------------------------------------------------
// State init: out[:,0,:] = 1, h frags = 0, xm frags = bf16(x[:,0,:]*1).
// A-frag mapping (16-bit, 16x32): M = lane&15, lh = lane>>4,
//   K = (e<8) ? e + 8*lh : e + 8 + 8*lh.
// ---------------------------------------------------------------------------
__global__ void init_state(const float* __restrict__ x, float* __restrict__ out,
                           unsigned short* __restrict__ hpk,
                           unsigned short* __restrict__ xmpk, unsigned* __restrict__ cnt) {
  int i = blockIdx.x * blockDim.x + threadIdx.x;
  if (i < B_ * F_) {
    int b = i >> 8, f = i & 255;
    out[(size_t)b * T_ * F_ + f] = 1.0f;     // m0 = ones at t=0
  }
  if (i < 2 * HPK_TILES * TILE_ELEMS) hpk[i] = 0;
  if (i < XMPK_TILES * TILE_ELEMS) {
    int e = i & 15, lane = (i >> 4) & 31, tix = i >> 9;
    int bi = tix >> 3, kt = tix & 7;
    int M = lane & 15, lh = lane >> 4;
    int K = (e < 8) ? (e + (lh << 3)) : (e + 8 + (lh << 3));
    int b = (bi << 4) + M, f = (kt << 5) + K;
    xmpk[i] = f2bf(x[(size_t)b * T_ * F_ + f]);   // t=0, m=1
  }
  if (i == 0) *cnt = 0u;
}

// ---------------------------------------------------------------------------
// Device-wide barrier: monotonic counter, one arrival per workgroup.
// ---------------------------------------------------------------------------
__device__ __forceinline__ void gbar(unsigned* cnt, unsigned target) {
  __syncthreads();
  if (threadIdx.x == 0) {
    __threadfence();
    __hip_atomic_fetch_add(cnt, 1u, __ATOMIC_ACQ_REL, __HIP_MEMORY_SCOPE_AGENT);
    while (__hip_atomic_load(cnt, __ATOMIC_RELAXED, __HIP_MEMORY_SCOPE_AGENT) < target)
      __builtin_amdgcn_s_sleep(2);
    __threadfence();
  }
  __syncthreads();
}

// ---------------------------------------------------------------------------
// Persistent recurrence kernel. Grid = 32 WGs x 256 threads (8 waves).
// wave (bi = wave-in-WG, hj = blockIdx) owns one 16x16 output tile of each
// gate for the whole run; h state stays in registers.
// ---------------------------------------------------------------------------
__global__ __launch_bounds__(256)
__attribute__((amdgpu_waves_per_eu(2)))
void gru_persistent(
    const float* __restrict__ x, const float* __restrict__ bz,
    const float* __restrict__ br, const float* __restrict__ bg,
    const float* __restrict__ b_out, float* __restrict__ out,
    const unsigned short* __restrict__ Wx, const unsigned short* __restrict__ Wh,
    const unsigned short* __restrict__ Wo, unsigned short* __restrict__ hpk,
    unsigned short* __restrict__ xmpk, unsigned* __restrict__ cnt) {
  extern __shared__ unsigned short smem[];

  const int lane = threadIdx.x & 31;
  const int bi = threadIdx.x >> 5;        // wave in WG -> batch tile 0..7
  const int lh = lane >> 4;
  const int N = lane & 15;
  const int hj = blockIdx.x;              // H column tile 0..31 (phase A)
  const int fj = blockIdx.x;              // F column tile (phase B, blocks<16)
  unsigned bar = 0;

  // ---- stage this WG's weight slices into LDS (once) -----------------------
  for (int i = threadIdx.x * 8; i < 24 * TILE_ELEMS; i += 256 * 8) {
    int tl = i >> 9, w = i & 511;                 // tl = kt*3+g
    int kt = tl / 3, g = tl % 3;
    *(uint4*)(&smem[SM_WX + i]) =
        *(const uint4*)(Wx + ((kt * 96 + g * 32 + hj) * TILE_ELEMS + w));
  }
  for (int i = threadIdx.x * 8; i < 48 * TILE_ELEMS; i += 256 * 8) {
    int tl = i >> 9, w = i & 511;
    int kt = tl / 3, g = tl % 3;
    *(uint4*)(&smem[SM_WH + i]) =
        *(const uint4*)(Wh + ((kt * 96 + g * 32 + hj) * TILE_ELEMS + w));
  }
  if (blockIdx.x < 16) {
    for (int i = threadIdx.x * 8; i < 16 * TILE_ELEMS; i += 256 * 8) {
      int tl = i >> 9, w = i & 511;
      *(uint4*)(&smem[SM_WO + i]) =
          *(const uint4*)(Wo + ((tl * 16 + fj) * TILE_ELEMS + w));
    }
  }
  __syncthreads();

  // ---- per-lane constants hoisted out of the time loop ---------------------
  const int jA = (hj << 4) + N;                   // H column (phase A)
  const float bzv = bz[jA], brv = br[jA], bgv = bg[jA];
  const int KA = jA & 31;
  const int epA = (KA & 7) | (((KA >> 4) & 1) << 3);
  const int lpbA = (((KA >> 3) & 1) << 4);        // lane offset from K
  const int tileA = bi * 16 + (jA >> 5);          // h-frag tile for scatter

  const int fB = (fj << 4) + N;                   // F column (phase B)
  const float bov = b_out[fB];
  const int KB = fB & 31;
  const int epB = (KB & 7) | (((KB >> 4) & 1) << 3);
  const int lpbB = (((KB >> 3) & 1) << 4);
  const int tileB = bi * 8 + (fB >> 5);           // xm-frag tile for scatter

  float hreg[8];                                  // recurrence state in VGPRs
  #pragma unroll
  for (int v = 0; v < 8; ++v) hreg[v] = 0.0f;

  for (int t = 0; t < T_ - 1; ++t) {
    asm volatile("" ::: "memory");                // no cross-step load motion
    const unsigned short* hrd = hpk + (t & 1) * (HPK_TILES * TILE_ELEMS);
    unsigned short* hwr = hpk + ((t & 1) ^ 1) * (HPK_TILES * TILE_ELEMS);

    // ---------------- Phase A: gates + h_new ----------------
    {
      v8f accZ = {0.f,0.f,0.f,0.f,0.f,0.f,0.f,0.f};
      v8f accR = accZ, accGA = accZ, accGB = accZ;
      // xm contribution (K over F)
      #pragma unroll 4
      for (int kt = 0; kt < F_ / 32; ++kt) {
        v16bf a  = load_frag_g(xmpk, bi * 8 + kt, lane);
        v16bf b0 = *(const v16bf*)(&smem[SM_WX + (kt * 3 + 0) * TILE_ELEMS + lane * 16]);
        v16bf b1 = *(const v16bf*)(&smem[SM_WX + (kt * 3 + 1) * TILE_ELEMS + lane * 16]);
        v16bf b2 = *(const v16bf*)(&smem[SM_WX + (kt * 3 + 2) * TILE_ELEMS + lane * 16]);
        accZ  = wmma_bf16(a, b0, accZ);
        accR  = wmma_bf16(a, b1, accR);
        accGA = wmma_bf16(a, b2, accGA);
      }
      // h contribution (K over H)
      #pragma unroll 4
      for (int kt = 0; kt < H_ / 32; ++kt) {
        v16bf a  = load_frag_g(hrd, bi * 16 + kt, lane);
        v16bf b0 = *(const v16bf*)(&smem[SM_WH + (kt * 3 + 0) * TILE_ELEMS + lane * 16]);
        v16bf b1 = *(const v16bf*)(&smem[SM_WH + (kt * 3 + 1) * TILE_ELEMS + lane * 16]);
        v16bf b2 = *(const v16bf*)(&smem[SM_WH + (kt * 3 + 2) * TILE_ELEMS + lane * 16]);
        accZ  = wmma_bf16(a, b0, accZ);
        accR  = wmma_bf16(a, b1, accR);
        accGB = wmma_bf16(a, b2, accGB);
      }
      // gate nonlinearities + recurrence update (C/D layout: M = v+8*lh)
      #pragma unroll
      for (int v = 0; v < 8; ++v) {
        int M = v + (lh << 3);
        float z = sigm(accZ[v] + bzv);
        float r = sigm(accR[v] + brv);
        float g = tanhf(accGA[v] + r * (accGB[v] + bgv));
        float hnew = (1.0f - z) * g + z * hreg[v];
        hreg[v] = hnew;
        hwr[(tileA * 32 + M + lpbA) * 16 + epA] = f2bf(hnew);
      }
    }
    gbar(cnt, ++bar * NWG);   // all h_new fragments visible

    // ---------------- Phase B: m = sigmoid(h_new @ Wout^T + b_out) -------
    if (blockIdx.x < 16) {
      v8f acc = {0.f,0.f,0.f,0.f,0.f,0.f,0.f,0.f};
      #pragma unroll 4
      for (int kt = 0; kt < H_ / 32; ++kt) {
        v16bf a = load_frag_g(hwr, bi * 16 + kt, lane);
        v16bf b = *(const v16bf*)(&smem[SM_WO + kt * TILE_ELEMS + lane * 16]);
        acc = wmma_bf16(a, b, acc);
      }
      #pragma unroll
      for (int v = 0; v < 8; ++v) {
        int M = v + (lh << 3);
        int bb = (bi << 4) + M;
        float mv = sigm(acc[v] + bov);
        size_t o = (size_t)bb * T_ * F_ + (size_t)(t + 1) * F_ + fB;
        out[o] = mv;
        float xn = x[o];                          // x[:, t+1, :]
        xmpk[(tileB * 32 + M + lpbB) * 16 + epB] = f2bf(xn * mv);
      }
    }
    gbar(cnt, ++bar * NWG);   // all m / xm fragments visible
  }
}

// ---------------------------------------------------------------------------
// Launch. Inputs (setup_inputs order):
// 0 inputs, 1 W_xz, 2 W_hz, 3 bz, 4 W_xr, 5 W_hr, 6 br,
// 7 W_xg, 8 W_hg, 9 bg, 10 W_out, 11 b_out.
// ---------------------------------------------------------------------------
extern "C" void kernel_launch(void* const* d_in, const int* in_sizes, int n_in,
                              void* d_out, int out_size, void* d_ws, size_t ws_size,
                              hipStream_t stream) {
  const float* x    = (const float*)d_in[0];
  const float* Wxz  = (const float*)d_in[1];
  const float* Whz  = (const float*)d_in[2];
  const float* bz   = (const float*)d_in[3];
  const float* Wxr  = (const float*)d_in[4];
  const float* Whr  = (const float*)d_in[5];
  const float* br   = (const float*)d_in[6];
  const float* Wxg  = (const float*)d_in[7];
  const float* Whg  = (const float*)d_in[8];
  const float* bg   = (const float*)d_in[9];
  const float* Wout = (const float*)d_in[10];
  const float* bo   = (const float*)d_in[11];
  float* out = (float*)d_out;

  uint8_t* ws = (uint8_t*)d_ws;
  unsigned short* Wx   = (unsigned short*)(ws + OFF_WX);
  unsigned short* Wh   = (unsigned short*)(ws + OFF_WH);
  unsigned short* Wo   = (unsigned short*)(ws + OFF_WO);
  unsigned short* hpk  = (unsigned short*)(ws + OFF_HPK);
  unsigned short* xmpk = (unsigned short*)(ws + OFF_XMPK);
  unsigned*       cnt  = (unsigned*)(ws + OFF_CNT);

  // 1) repack weights to bf16 WMMA fragments (L2-resident, 2.6 MB)
  {
    int n = (WX_TILES + WH_TILES + WO_TILES) * TILE_ELEMS;   // 1,310,720
    pack_weights<<<(n + 255) / 256, 256, 0, stream>>>(
        Wxz, Wxr, Wxg, Whz, Whr, Whg, Wout, Wx, Wh, Wo);
  }
  // 2) init recurrence state + t=0 output row
  {
    int n = 2 * HPK_TILES * TILE_ELEMS;                      // largest range
    init_state<<<(n + 255) / 256, 256, 0, stream>>>(x, out, hpk, xmpk, cnt);
  }
  // 3) persistent recurrence: 32 WGs x 256 threads, 88KB LDS weight cache,
  //    2 global barriers / step
  gru_persistent<<<NWG, 256, SM_BYTES, stream>>>(
      x, bz, br, bg, bo, out, Wx, Wh, Wo, hpk, xmpk, cnt);
}
